// SpatialContext_42949673368
// MI455X (gfx1250) — matrix operand order repaired and checked by reading, not compiled, not run on hardware
//
#include <hip/hip_runtime.h>
#include <hip/hip_bf16.h>

typedef __attribute__((ext_vector_type(16))) _Float16     v16h;
typedef __attribute__((ext_vector_type(8)))  float        v8f;
typedef __attribute__((ext_vector_type(8)))  unsigned int v8u;
typedef __attribute__((ext_vector_type(4)))  int          v4i_;

typedef __attribute__((address_space(1))) v4i_* glb_v4i;
typedef __attribute__((address_space(3))) v4i_* lds_v4i;
typedef __attribute__((address_space(1))) void* glb_void;
typedef __attribute__((address_space(3))) void* lds_void;

#define SCALEF 0.25f   // (hd=16)^-0.5

#if __has_builtin(__builtin_amdgcn_global_load_async_to_lds_b128)
#define HAVE_ASYNC_LDS 1
#else
#define HAVE_ASYNC_LDS 0
#endif

// ---------------------------------------------------------------------------
// WMMA helpers (gfx1250, wave32). D = A(16x32 f16) * B(32x16 f16) + C(16x16 f32)
// ---------------------------------------------------------------------------
__device__ inline v8f wmma_f16(v16h a, v16h b, v8f c) {
  return __builtin_amdgcn_wmma_f32_16x16x32_f16(false, a, false, b, (short)0, c,
                                                false, false);
}

// B fragment from pre-packed tile buffer: word layout [lane(32)][r(8)] so each
// lane issues two coalesced global_load_b128.
__device__ inline v16h load_b_frag(const unsigned int* __restrict__ pk, int lane) {
  const uint4* p = reinterpret_cast<const uint4*>(pk) + lane * 2;
  uint4 lo = p[0], hi = p[1];
  v8u w;
  w[0] = lo.x; w[1] = lo.y; w[2] = lo.z; w[3] = lo.w;
  w[4] = hi.x; w[5] = hi.y; w[6] = hi.z; w[7] = hi.w;
  return __builtin_bit_cast(v16h, w);
}

// A fragment gather from an LDS f16 array, row-major [M][ldk].
// A-layout (matches packed B): lane<16: VGPR0-7 K=0..15; lane>=16: K=16..31.
__device__ inline v16h load_a_frag_lds(const _Float16* sA, int ldk, int mbase,
                                       int koff, int lane) {
  int m = mbase + (lane & 15), half = lane >> 4;
  v8u w;
#pragma unroll
  for (int r = 0; r < 8; ++r) {
    int k = koff + half * 16 + 2 * r;
    w[r] = *reinterpret_cast<const unsigned int*>(&sA[m * ldk + k]);
  }
  return __builtin_bit_cast(v16h, w);
}

// One 16B async (ASYNCcnt-tracked) DMA transfer: global -> LDS.
__device__ inline void async_copy_b128(const void* g, void* l) {
#if HAVE_ASYNC_LDS
  __builtin_amdgcn_global_load_async_to_lds_b128(
      (glb_v4i)(glb_void)const_cast<void*>(g), (lds_v4i)(lds_void)l, 0, 0);
#endif
}

__device__ inline void wait_async_zero() {
#if HAVE_ASYNC_LDS
#if __has_builtin(__builtin_amdgcn_s_wait_asynccnt)
  __builtin_amdgcn_s_wait_asynccnt(0);
#else
  asm volatile("s_wait_asynccnt 0x0" ::: "memory");
#endif
#endif
}

// Stage a contiguous tile (n_vec * 16B) from global into LDS.
__device__ inline void stage_tile(const uint4* __restrict__ src, uint4* dst,
                                  int n_vec, int tid, int nthreads) {
#if HAVE_ASYNC_LDS
  for (int i = tid; i < n_vec; i += nthreads) async_copy_b128(src + i, dst + i);
  wait_async_zero();
#else
  for (int i = tid; i < n_vec; i += nthreads) dst[i] = src[i];
#endif
}

// ---------------------------------------------------------------------------
// k0: pack f32 weights into f16 WMMA-B tile layout.
// tile index = nt*KT + kt ; word = [lane][r] ; k0 = kt*32 + half*16 + 2r
// mode 0: B[k][n] = src[n*K + k]            (W.T of row-major (N,K) weight)
// mode 1: B[k][n] = conv_w[n*576 + (k%64)*9 + k/64]   (k = (ki*3+kj)*64 + c)
// ---------------------------------------------------------------------------
__global__ void k0_pack(const float* __restrict__ src,
                        unsigned int* __restrict__ dst, int K, int N, int mode) {
  int w = blockIdx.x * 256 + threadIdx.x;
  int KT = K >> 5;
  int total = (KT * (N >> 4)) << 8;
  if (w >= total) return;
  int tile = w >> 8, wi = w & 255, lane = wi >> 3, r = wi & 7;
  int nt = tile / KT, kt = tile % KT;
  int nn = lane & 15, half = lane >> 4;
  int k0 = kt * 32 + half * 16 + 2 * r;
  int n = nt * 16 + nn;
  float f0, f1;
  if (mode == 0) {
    f0 = src[(size_t)n * K + k0];
    f1 = src[(size_t)n * K + k0 + 1];
  } else {
    f0 = src[(size_t)n * 576 + (k0 & 63) * 9 + (k0 >> 6)];
    f1 = src[(size_t)n * 576 + ((k0 + 1) & 63) * 9 + ((k0 + 1) >> 6)];
  }
  _Float16 h0 = (_Float16)f0, h1 = (_Float16)f1;
  unsigned int u0 = (unsigned int)__builtin_bit_cast(unsigned short, h0);
  unsigned int u1 = (unsigned int)__builtin_bit_cast(unsigned short, h1);
  dst[w] = u0 | (u1 << 16);
}

// ---------------------------------------------------------------------------
// k1: one wave per 3x3 window. LN -> QKV (WMMA) -> attention (VALU, 9x9) ->
// 3x3 conv via implicit GEMM (WMMA, K=576). Writes conv out (f16) at un-
// windowed location: buf1[(b*576 + l)*128 + oc]. Rows m>=9 go to dump slots
// (branchless epilogues; keeps EXEC untouched around WMMA).
// ---------------------------------------------------------------------------
struct WaveScratch {
  float    raw[576];       // x window, token-major [9][64]
  _Float16 aln[1024];      // LN'd, padded to 16 rows [16][64]
  float    qkvs[10 * 192]; // q|k|v per token, row 9 = dump
  _Float16 os[640];        // attn out in CONV channel order [9][64] (c=head*16+d)
  float    mu[9];
  float    rsig[9];
};

__global__ void __launch_bounds__(128)
k1_win(const float* __restrict__ x, const float* __restrict__ qkv_b,
       const float* __restrict__ rel_table, const float* __restrict__ n1g,
       const float* __restrict__ n1b, const float* __restrict__ conv_b,
       const unsigned int* __restrict__ qkvw_pk,
       const unsigned int* __restrict__ convw_pk, _Float16* __restrict__ buf1,
       size_t dump_off) {
  __shared__ WaveScratch smem[4];
  int wave = threadIdx.x >> 5, lane = threadIdx.x & 31;
  WaveScratch& S = smem[wave];
  int gw = blockIdx.x * 4 + wave; // window id in [0, 32768)
  int b = gw >> 6, nwin = gw & 63, u = nwin >> 3, v = nwin & 7;
  const float* xb = x + (size_t)b * 64 * 576;

  // ---- load window: token t=(iy,ix), channel c ----
  for (int idx = lane; idx < 576; idx += 32) {
    int t = idx >> 6, c = idx & 63;
    int hh = u * 3 + t / 3, ww = v * 3 + t % 3;
    S.raw[idx] = xb[(size_t)c * 576 + hh * 24 + ww];
  }
  __syncthreads();

  // ---- LayerNorm per token over 64 channels ----
  if (lane < 9) {
    float s = 0.f, s2 = 0.f;
    for (int c = 0; c < 64; ++c) { float f = S.raw[lane * 64 + c]; s += f; s2 += f * f; }
    float m = s * (1.f / 64.f);
    S.mu[lane] = m;
    S.rsig[lane] = rsqrtf(s2 * (1.f / 64.f) - m * m + 1e-5f);
  }
  __syncthreads();
  for (int idx = lane; idx < 1024; idx += 32) {
    int t = idx >> 6, c = idx & 63;
    float vv = 0.f;
    if (t < 9) vv = (S.raw[idx] - S.mu[t]) * S.rsig[t] * n1g[c] + n1b[c];
    S.aln[idx] = (_Float16)vv;
  }
  __syncthreads();

  // ---- QKV: [16x64] x [64x192] WMMA ----
  int nn = lane & 15, half = lane >> 4;
  v16h a0 = load_a_frag_lds(S.aln, 64, 0, 0, lane);
  v16h a1 = load_a_frag_lds(S.aln, 64, 0, 32, lane);
#pragma unroll
  for (int nt = 0; nt < 12; ++nt) {
    v8f acc = {};
    acc = wmma_f16(a0, load_b_frag(qkvw_pk + (nt * 2 + 0) * 256, lane), acc);
    acc = wmma_f16(a1, load_b_frag(qkvw_pk + (nt * 2 + 1) * 256, lane), acc);
    int col = nt * 16 + nn;
    float bb = qkv_b[col];
#pragma unroll
    for (int g = 0; g < 8; ++g) {
      int m = g + 8 * half;
      S.qkvs[(m < 9 ? m : 9) * 192 + col] = acc[g] + bb; // row 9 = dump
    }
  }
  __syncthreads();

  // ---- attention: 4 heads x 9 queries = 36 rows ----
  for (int t2 = lane; t2 < 36; t2 += 32) {
    int head = t2 & 3, i = t2 >> 2;
    int oyi = i / 3, oxi = i % 3;
    float sc[9], mx = -1e30f;
    for (int j = 0; j < 9; ++j) {
      float d = 0.f;
      for (int dd = 0; dd < 16; ++dd)
        d += S.qkvs[i * 192 + dd * 4 + head] * S.qkvs[j * 192 + 64 + dd * 4 + head];
      int ci = oyi - j / 3 + 2, cj = oxi - j % 3 + 2;
      d = d * SCALEF + rel_table[(ci * 5 + cj) * 4 + head];
      sc[j] = d; mx = fmaxf(mx, d);
    }
    float ssum = 0.f;
    for (int j = 0; j < 9; ++j) { sc[j] = __expf(sc[j] - mx); ssum += sc[j]; }
    float inv = 1.f / ssum;
    for (int dd = 0; dd < 16; ++dd) {
      float o = 0.f;
      for (int j = 0; j < 9; ++j)
        o += sc[j] * S.qkvs[j * 192 + 128 + dd * 4 + head];
      // conv channel order: c = head*16 + dd
      S.os[i * 64 + head * 16 + dd] = (_Float16)(o * inv);
    }
  }
  __syncthreads();

  // ---- conv as implicit GEMM: M=16(9 used), N=128, K=576 ----
  v8f acc[8];
#pragma unroll
  for (int t = 0; t < 8; ++t) acc[t] = (v8f){};
  int p = nn; // output position (row of A)
  for (int kt = 0; kt < 18; ++kt) {
    v8u aw;
#pragma unroll
    for (int r = 0; r < 8; ++r) {
      int k0 = kt * 32 + half * 16 + 2 * r;
      int t3 = k0 >> 6, c = k0 & 63;       // t3 = ki*3+kj
      int ki = t3 / 3, kj = t3 % 3;
      int iy = p / 3 + ki - 1, ix = p % 3 + kj - 1;
      unsigned int w = 0u;
      if (p < 9 && iy >= 0 && iy < 3 && ix >= 0 && ix < 3)
        w = *reinterpret_cast<const unsigned int*>(&S.os[(iy * 3 + ix) * 64 + c]);
      aw[r] = w;
    }
    v16h af = __builtin_bit_cast(v16h, aw);
#pragma unroll
    for (int nt = 0; nt < 8; ++nt)
      acc[nt] = wmma_f16(af, load_b_frag(convw_pk + (nt * 18 + kt) * 256, lane),
                         acc[nt]);
  }
#pragma unroll
  for (int nt = 0; nt < 8; ++nt) {
    int col = nt * 16 + nn;
    float cb = conv_b[col];
#pragma unroll
    for (int g = 0; g < 8; ++g) {
      int pp = g + 8 * half;
      int py = pp / 3, px = pp - py * 3;
      size_t l = (size_t)(u * 3 + py) * 24 + (v * 3 + px);
      size_t idx = (pp < 9) ? (((size_t)b * 576 + l) * 128 + col)
                            : (dump_off + col);
      buf1[idx] = (_Float16)(acc[nt][g] + cb);
    }
  }
}

// ---------------------------------------------------------------------------
// k2: proj GEMM [294912 x 128] x [128 x 128] + bias -> f32 buf2
// ---------------------------------------------------------------------------
__global__ void __launch_bounds__(256)
k2_proj(const _Float16* __restrict__ Ain, const unsigned int* __restrict__ Bpk,
        const float* __restrict__ bias, float* __restrict__ Cout) {
  __shared__ __align__(16) _Float16 sA[128 * 128];
  int tid = threadIdx.x;
  size_t m0 = (size_t)blockIdx.x * 128;
  stage_tile(reinterpret_cast<const uint4*>(Ain + m0 * 128),
             reinterpret_cast<uint4*>(sA), 128 * 128 / 8, tid, 256);
  __syncthreads();
  int wave = tid >> 5, lane = tid & 31, nn = lane & 15, half = lane >> 4;
  int mbase = wave * 16;
  v16h a[4];
#pragma unroll
  for (int kt = 0; kt < 4; ++kt) a[kt] = load_a_frag_lds(sA, 128, mbase, kt * 32, lane);
#pragma unroll
  for (int nt = 0; nt < 8; ++nt) {
    v8f acc = {};
#pragma unroll
    for (int kt = 0; kt < 4; ++kt)
      acc = wmma_f16(a[kt], load_b_frag(Bpk + (nt * 4 + kt) * 256, lane), acc);
    int col = nt * 16 + nn;
    float bb = bias[col];
#pragma unroll
    for (int g = 0; g < 8; ++g) {
      size_t row = m0 + mbase + g + 8 * half;
      Cout[row * 128 + col] = acc[g] + bb;
    }
  }
}

// ---------------------------------------------------------------------------
// k3: LN(o) -> fc1 [294912 x 128] x [128 x 256] + bias -> exact GELU -> f16
// ---------------------------------------------------------------------------
__global__ void __launch_bounds__(256)
k3_fc1(const float* __restrict__ O, const unsigned int* __restrict__ Bpk,
       const float* __restrict__ bias, const float* __restrict__ g2,
       const float* __restrict__ b2, _Float16* __restrict__ Hout) {
  __shared__ __align__(16) _Float16 sA[128 * 128];
  int tid = threadIdx.x;
  size_t m0 = (size_t)blockIdx.x * 128;
  if (tid < 128) {
    const float4* rowp = reinterpret_cast<const float4*>(O + (m0 + tid) * 128);
    float s = 0.f, s2 = 0.f;
    float4 buf[32];
#pragma unroll
    for (int q = 0; q < 32; ++q) {
      float4 f = rowp[q];
      buf[q] = f;
      s += f.x + f.y + f.z + f.w;
      s2 += f.x * f.x + f.y * f.y + f.z * f.z + f.w * f.w;
    }
    float mean = s * (1.f / 128.f);
    float rs = rsqrtf(s2 * (1.f / 128.f) - mean * mean + 1e-5f);
#pragma unroll
    for (int q = 0; q < 32; ++q) {
      float4 f = buf[q];
      int c = q * 4;
      sA[tid * 128 + c + 0] = (_Float16)((f.x - mean) * rs * g2[c + 0] + b2[c + 0]);
      sA[tid * 128 + c + 1] = (_Float16)((f.y - mean) * rs * g2[c + 1] + b2[c + 1]);
      sA[tid * 128 + c + 2] = (_Float16)((f.z - mean) * rs * g2[c + 2] + b2[c + 2]);
      sA[tid * 128 + c + 3] = (_Float16)((f.w - mean) * rs * g2[c + 3] + b2[c + 3]);
    }
  }
  __syncthreads();
  int wave = tid >> 5, lane = tid & 31, nn = lane & 15, half = lane >> 4;
  int mbase = wave * 16;
  v16h a[4];
#pragma unroll
  for (int kt = 0; kt < 4; ++kt) a[kt] = load_a_frag_lds(sA, 128, mbase, kt * 32, lane);
#pragma unroll
  for (int nt = 0; nt < 16; ++nt) {
    v8f acc = {};
#pragma unroll
    for (int kt = 0; kt < 4; ++kt)
      acc = wmma_f16(a[kt], load_b_frag(Bpk + (nt * 4 + kt) * 256, lane), acc);
    int col = nt * 16 + nn;
    float bb = bias[col];
#pragma unroll
    for (int g = 0; g < 8; ++g) {
      size_t row = m0 + mbase + g + 8 * half;
      float vv = acc[g] + bb;
      vv = 0.5f * vv * (1.f + erff(vv * 0.70710678118f)); // exact GELU
      Hout[row * 256 + col] = (_Float16)vv;
    }
  }
}

// ---------------------------------------------------------------------------
// k4: fc2 [294912 x 256] x [256 x 128] + bias + residual(o) ->
// transpose-store to (B, 128, 24, 24). K processed in two 128-chunks (32KB LDS)
// ---------------------------------------------------------------------------
__global__ void __launch_bounds__(256)
k4_fc2(const _Float16* __restrict__ Hin, const unsigned int* __restrict__ Bpk,
       const float* __restrict__ bias, const float* __restrict__ O,
       float* __restrict__ out) {
  __shared__ __align__(16) _Float16 sA[128 * 128];
  int tid = threadIdx.x;
  size_t m0 = (size_t)blockIdx.x * 128;
  int wave = tid >> 5, lane = tid & 31, nn = lane & 15, half = lane >> 4;
  int mbase = wave * 16;
  const uint4* src = reinterpret_cast<const uint4*>(Hin + m0 * 256);
  uint4* dst = reinterpret_cast<uint4*>(sA);
  v8f acc[8];
#pragma unroll
  for (int t = 0; t < 8; ++t) acc[t] = (v8f){};
  for (int ch = 0; ch < 2; ++ch) {
#if HAVE_ASYNC_LDS
    for (int i = tid; i < 128 * 16; i += 256) {
      int rrow = i >> 4, seg = i & 15;
      async_copy_b128(src + rrow * 32 + ch * 16 + seg, dst + rrow * 16 + seg);
    }
    wait_async_zero();
#else
    for (int i = tid; i < 128 * 16; i += 256) {
      int rrow = i >> 4, seg = i & 15;
      dst[rrow * 16 + seg] = src[rrow * 32 + ch * 16 + seg];
    }
#endif
    __syncthreads();
    v16h a[4];
#pragma unroll
    for (int kt = 0; kt < 4; ++kt) a[kt] = load_a_frag_lds(sA, 128, mbase, kt * 32, lane);
#pragma unroll
    for (int nt = 0; nt < 8; ++nt)
#pragma unroll
      for (int kt = 0; kt < 4; ++kt)
        acc[nt] = wmma_f16(a[kt],
                           load_b_frag(Bpk + (nt * 8 + ch * 4 + kt) * 256, lane),
                           acc[nt]);
    __syncthreads();
  }
#pragma unroll
  for (int nt = 0; nt < 8; ++nt) {
    int col = nt * 16 + nn;
    float bb = bias[col];
#pragma unroll
    for (int g = 0; g < 8; ++g) {
      size_t row = m0 + mbase + g + 8 * half;
      float y = acc[nt][g] + bb + O[row * 128 + col];
      size_t bidx = row / 576, l = row % 576;
      size_t hh = l / 24, ww = l % 24;
      out[((bidx * 128 + col) * 24 + hh) * 24 + ww] = y;
    }
  }
}

// ---------------------------------------------------------------------------
extern "C" void kernel_launch(void* const* d_in, const int* in_sizes, int n_in,
                              void* d_out, int out_size, void* d_ws, size_t ws_size,
                              hipStream_t stream) {
  const float* x         = (const float*)d_in[0];
  const float* qkv_w     = (const float*)d_in[1];
  const float* qkv_b     = (const float*)d_in[2];
  const float* rel_table = (const float*)d_in[3];
  const float* proj_w    = (const float*)d_in[4];
  const float* proj_b    = (const float*)d_in[5];
  const float* fc1_w     = (const float*)d_in[6];
  const float* fc1_b     = (const float*)d_in[7];
  const float* fc2_w     = (const float*)d_in[8];
  const float* fc2_b     = (const float*)d_in[9];
  const float* n1g       = (const float*)d_in[10];
  const float* n1b       = (const float*)d_in[11];
  const float* n2g       = (const float*)d_in[12];
  const float* n2b       = (const float*)d_in[13];
  const float* conv_w    = (const float*)d_in[14];
  const float* conv_b    = (const float*)d_in[15];

  size_t off = 0;
  auto alloc = [&](size_t bytes) {
    void* p = (char*)d_ws + off;
    off += (bytes + 255) & ~(size_t)255;
    return p;
  };
  unsigned int* qkvw_pk  = (unsigned int*)alloc((size_t)24 * 256 * 4);
  unsigned int* convw_pk = (unsigned int*)alloc((size_t)144 * 256 * 4);
  unsigned int* projw_pk = (unsigned int*)alloc((size_t)32 * 256 * 4);
  unsigned int* fc1w_pk  = (unsigned int*)alloc((size_t)64 * 256 * 4);
  unsigned int* fc2w_pk  = (unsigned int*)alloc((size_t)64 * 256 * 4);
  const size_t ROWS = (size_t)512 * 576; // 294912
  float*    buf2 = (float*)alloc(ROWS * 128 * 4);            // proj out (f32)
  _Float16* buf1 = (_Float16*)alloc((ROWS * 128 + 256) * 2); // conv out + dump
  _Float16* buf3 = (_Float16*)alloc(ROWS * 256 * 2);         // gelu(fc1) (f16)
  const size_t dump_off = ROWS * 128;

  // weight packing (tiny)
  k0_pack<<<24, 256, 0, stream>>>(qkv_w,  qkvw_pk,  64, 192, 0);
  k0_pack<<<144, 256, 0, stream>>>(conv_w, convw_pk, 576, 128, 1);
  k0_pack<<<32, 256, 0, stream>>>(proj_w, projw_pk, 128, 128, 0);
  k0_pack<<<64, 256, 0, stream>>>(fc1_w,  fc1w_pk,  128, 256, 0);
  k0_pack<<<64, 256, 0, stream>>>(fc2_w,  fc2w_pk,  256, 128, 0);

  // fused LN + QKV + attention + conv  (32768 windows, 1 wave each)
  k1_win<<<8192, 128, 0, stream>>>(x, qkv_b, rel_table, n1g, n1b, conv_b,
                                   qkvw_pk, convw_pk, buf1, dump_off);
  // proj
  k2_proj<<<2304, 256, 0, stream>>>(buf1, projw_pk, proj_b, buf2);
  // LN + fc1 + GELU
  k3_fc1<<<2304, 256, 0, stream>>>(buf2, fc1w_pk, fc1_b, n2g, n2b, buf3);
  // fc2 + residual + transpose out
  k4_fc2<<<2304, 256, 0, stream>>>(buf3, fc2w_pk, fc2_b, buf2, (float*)d_out);
}
